// NODELSTMCell_78185584656869
// MI455X (gfx1250) — matrix operand order repaired and checked by reading, not compiled, not run on hardware
//
#include <hip/hip_runtime.h>
#include <hip/hip_bf16.h>

#define BB 8192
#define II 256
#define HH 1024

#define LSTRIDE 40   // padded bf16 elements per staged W row (32 K + 8 pad -> 80B pitch)
#define LROWS   64   // W rows staged per block per K-step

typedef __bf16 bf16_t;
typedef __attribute__((ext_vector_type(16))) __bf16 v16bf;
typedef __attribute__((ext_vector_type(8)))  __bf16 v8bf;
typedef __attribute__((ext_vector_type(8)))  float   v8f;

// ---------------------------------------------------------------------------
// WMMA fragment helpers (rows/cols per CDNA5 ISA §7.12.2, wave32; K order contiguous, not ISA table)
// A 16x32 bf16: lane L holds row M=L%16; khalf=L/16.
//   elements 0..15 -> K = khalf*16 + e (contiguous, matches B)
// B 32x16 bf16 (W^T with W row-major [N,K]): lane L holds col N=L%16;
//   elements 0..15 -> K = (L/16)*16 + e  (16 contiguous bf16 in W row N)
// ---------------------------------------------------------------------------
static __device__ __forceinline__ v16bf load_a_frag(const bf16_t* __restrict__ A,
                                                    int lda, int row, int k0, int khalf) {
  const bf16_t* p = A + (size_t)row * lda + k0 + khalf * 16;
  v8bf lo = *reinterpret_cast<const v8bf*>(p);
  v8bf hi = *reinterpret_cast<const v8bf*>(p + 8);
  v16bf r;
#pragma unroll
  for (int i = 0; i < 8; ++i) { r[i] = lo[i]; r[i + 8] = hi[i]; }
  return r;
}

// one 32B contiguous global load of a W row k-slice half (16 bf16, 32B aligned)
static __device__ __forceinline__ v16bf load_wslice(const bf16_t* __restrict__ W,
                                                    int ldw, int wrow, int k0, int half) {
  return *reinterpret_cast<const v16bf*>(W + (size_t)wrow * ldw + k0 + half * 16);
}

static __device__ __forceinline__ void stage_store(bf16_t* sbuf, int r, int half, v16bf v) {
  v8bf lo, hi;
#pragma unroll
  for (int i = 0; i < 8; ++i) { lo[i] = v[i]; hi[i] = v[i + 8]; }
  bf16_t* p = sbuf + r * LSTRIDE + half * 16;
  *reinterpret_cast<v8bf*>(p)     = lo;
  *reinterpret_cast<v8bf*>(p + 8) = hi;
}

static __device__ __forceinline__ v16bf lds_b_frag(const bf16_t* sbuf, int c, int khalf) {
  const bf16_t* p = sbuf + c * LSTRIDE + khalf * 16;
  v8bf lo = *reinterpret_cast<const v8bf*>(p);
  v8bf hi = *reinterpret_cast<const v8bf*>(p + 8);
  v16bf r;
#pragma unroll
  for (int i = 0; i < 8; ++i) { r[i] = lo[i]; r[i + 8] = hi[i]; }
  return r;
}

static __device__ __forceinline__ float sigmoidf_(float x) {
  return 1.0f / (1.0f + __expf(-x));
}

#define WMMA_BF16(A_, B_, C_) \
  __builtin_amdgcn_wmma_f32_16x16x32_bf16(false, (A_), false, (B_), (short)0, (C_), false, false)

// ---------------------------------------------------------------------------
// Conversion / preparation kernels
// ---------------------------------------------------------------------------
__global__ void k_cvt_bf16(const float* __restrict__ src, bf16_t* __restrict__ dst, int n) {
  int i = blockIdx.x * blockDim.x + threadIdx.x;
  int stride = gridDim.x * blockDim.x;
  for (; i < n; i += stride) dst[i] = (bf16_t)src[i];
}

// W1 is [H, H+1]; keep first H columns as bf16 [H, H] row-major
__global__ void k_cvt_W1(const float* __restrict__ W1, bf16_t* __restrict__ dst) {
  int i = blockIdx.x * blockDim.x + threadIdx.x;
  int stride = gridDim.x * blockDim.x;
  const int n = HH * HH;
  for (; i < n; i += stride) {
    int r = i / HH, c = i % HH;
    dst[i] = (bf16_t)W1[(size_t)r * (HH + 1) + c];
  }
}

// bsum = b_ih + b_hh;  b1_eff[s] = b1 + t_s * W1[:, H]  for t_s in {t0, t0+dt/2, t0+dt}
__global__ void k_bias(const float* __restrict__ b_ih, const float* __restrict__ b_hh,
                       const float* __restrict__ b1, const float* __restrict__ W1,
                       const float* __restrict__ t0p, const float* __restrict__ t1p,
                       float* __restrict__ bsum, float* __restrict__ b1eff) {
  int i = blockIdx.x * blockDim.x + threadIdx.x;
  if (i < 4 * HH) bsum[i] = b_ih[i] + b_hh[i];
  if (i < HH) {
    float w  = W1[(size_t)i * (HH + 1) + HH];
    float t0 = t0p[0];
    float dt = t1p[0] - t0p[0];
    float b  = b1[i];
    b1eff[0 * HH + i] = b + t0 * w;
    b1eff[1 * HH + i] = b + (t0 + 0.5f * dt) * w;
    b1eff[2 * HH + i] = b + (t0 + dt) * w;
  }
}

// ---------------------------------------------------------------------------
// Fused LSTM-gates GEMM. Block = 4 waves; all waves share one 16-col block n0
// and the same 64 staged W rows (4 gates x 16 cols) per K-step via LDS.
// Each wave owns 32 rows (2 M-subtiles) x 16 cols x 4 gates = 8 accumulators.
// ---------------------------------------------------------------------------
__global__ __launch_bounds__(128) void k_gates(
    const bf16_t* __restrict__ Xb, const bf16_t* __restrict__ Hb,
    const bf16_t* __restrict__ Wih, const bf16_t* __restrict__ Whh,
    const float* __restrict__ bsum, const float* __restrict__ c_in,
    float* __restrict__ c_out, float* __restrict__ ht32, bf16_t* __restrict__ htb) {
  __shared__ __align__(32) bf16_t smem[2][LROWS * LSTRIDE];

  const int tid  = threadIdx.x;
  const int wv   = tid >> 5;
  const int lane = tid & 31;
  const int nl   = lane & 15;
  const int kh   = lane >> 4;

  const int TN   = HH / 16;                       // 64 column tiles
  const int n0   = (blockIdx.x % TN) * 16;
  const int m0w  = (blockIdx.x / TN) * 128 + wv * 32;
  const int rowA0 = m0w + nl;
  const int rowA1 = m0w + 16 + nl;

  // staging thread mapping: thread t loads W row sr (0..63), k-half sh
  const int sr = tid >> 1, sh = tid & 1;
  const int wrow = (sr >> 4) * HH + n0 + (sr & 15);  // gate (sr>>4), col n0+(sr&15)

  v8f acc[2][4] = {};

  // ---- phase 1: x @ W_ih^T  (K = II) ----
  {
    v16bf breg = load_wslice(Wih, II, wrow, 0, sh);
    stage_store(smem[0], sr, sh, breg);
    __syncthreads();
    for (int i = 0; i < II / 32; ++i) {
      const int cur = i & 1;
      const bool more = (i + 1) < (II / 32);
      if (more) breg = load_wslice(Wih, II, wrow, (i + 1) * 32, sh);
      const int k0 = i * 32;
      v16bf a0 = load_a_frag(Xb, II, rowA0, k0, kh);
      v16bf a1 = load_a_frag(Xb, II, rowA1, k0, kh);
#pragma unroll
      for (int g = 0; g < 4; ++g) {
        v16bf b = lds_b_frag(smem[cur], g * 16 + nl, kh);
        acc[0][g] = WMMA_BF16(a0, b, acc[0][g]);
        acc[1][g] = WMMA_BF16(a1, b, acc[1][g]);
      }
      if (more) stage_store(smem[cur ^ 1], sr, sh, breg);
      __syncthreads();
    }
  }
  // ---- phase 2: h @ W_hh^T  (K = HH) ----
  {
    v16bf breg = load_wslice(Whh, HH, wrow, 0, sh);
    stage_store(smem[0], sr, sh, breg);
    __syncthreads();
    for (int i = 0; i < HH / 32; ++i) {
      const int cur = i & 1;
      const bool more = (i + 1) < (HH / 32);
      if (more) breg = load_wslice(Whh, HH, wrow, (i + 1) * 32, sh);
      const int k0 = i * 32;
      v16bf a0 = load_a_frag(Hb, HH, rowA0, k0, kh);
      v16bf a1 = load_a_frag(Hb, HH, rowA1, k0, kh);
#pragma unroll
      for (int g = 0; g < 4; ++g) {
        v16bf b = lds_b_frag(smem[cur], g * 16 + nl, kh);
        acc[0][g] = WMMA_BF16(a0, b, acc[0][g]);
        acc[1][g] = WMMA_BF16(a1, b, acc[1][g]);
      }
      if (more) stage_store(smem[cur ^ 1], sr, sh, breg);
      __syncthreads();
    }
  }

  // ---- epilogue: LSTM pointwise. C/D layout: VGPR r, lane l -> M=r+8*(l>>4), N=l%16
  const int n = n0 + nl;
  const float bi = bsum[0 * HH + n];
  const float bf = bsum[1 * HH + n];
  const float bg = bsum[2 * HH + n];
  const float bo = bsum[3 * HH + n];
#pragma unroll
  for (int ms = 0; ms < 2; ++ms) {
#pragma unroll
    for (int r = 0; r < 8; ++r) {
      int m = m0w + ms * 16 + kh * 8 + r;
      size_t idx = (size_t)m * HH + n;
      float gi = acc[ms][0][r] + bi;
      float gf = acc[ms][1][r] + bf;
      float gg = acc[ms][2][r] + bg;
      float go = acc[ms][3][r] + bo;
      float cn = sigmoidf_(gf) * c_in[idx] + sigmoidf_(gi) * tanhf(gg);
      float ht = sigmoidf_(go) * tanhf(cn);
      c_out[idx] = cn;
      ht32[idx]  = ht;
      htb[idx]   = (bf16_t)ht;
    }
  }
}

// ---------------------------------------------------------------------------
// f_node layer 1: Z = tanh(A @ W1b^T + b1_eff), bf16 out.
// Block = 4 waves sharing 64 staged W rows (cols ng..ng+63); wave = 32x64 tile.
// ---------------------------------------------------------------------------
__global__ __launch_bounds__(128) void k_fnode_l1(
    const bf16_t* __restrict__ Ab, const bf16_t* __restrict__ W1b,
    const float* __restrict__ b1eff, bf16_t* __restrict__ Zb) {
  __shared__ __align__(32) bf16_t smem[2][LROWS * LSTRIDE];

  const int tid  = threadIdx.x;
  const int wv   = tid >> 5;
  const int lane = tid & 31;
  const int nl   = lane & 15;
  const int kh   = lane >> 4;

  const int TG   = HH / 64;                        // 16 column groups
  const int ng   = (blockIdx.x % TG) * 64;
  const int m0w  = (blockIdx.x / TG) * 128 + wv * 32;
  const int rowA0 = m0w + nl;
  const int rowA1 = m0w + 16 + nl;

  const int sr = tid >> 1, sh = tid & 1;
  const int wrow = ng + sr;

  v8f acc[2][4] = {};
  v16bf breg = load_wslice(W1b, HH, wrow, 0, sh);
  stage_store(smem[0], sr, sh, breg);
  __syncthreads();
  for (int i = 0; i < HH / 32; ++i) {
    const int cur = i & 1;
    const bool more = (i + 1) < (HH / 32);
    if (more) breg = load_wslice(W1b, HH, wrow, (i + 1) * 32, sh);
    const int k0 = i * 32;
    v16bf a0 = load_a_frag(Ab, HH, rowA0, k0, kh);
    v16bf a1 = load_a_frag(Ab, HH, rowA1, k0, kh);
#pragma unroll
    for (int j = 0; j < 4; ++j) {
      v16bf b = lds_b_frag(smem[cur], j * 16 + nl, kh);
      acc[0][j] = WMMA_BF16(a0, b, acc[0][j]);
      acc[1][j] = WMMA_BF16(a1, b, acc[1][j]);
    }
    if (more) stage_store(smem[cur ^ 1], sr, sh, breg);
    __syncthreads();
  }

#pragma unroll
  for (int j = 0; j < 4; ++j) {
    int n = ng + j * 16 + nl;
    float bias = b1eff[n];
#pragma unroll
    for (int ms = 0; ms < 2; ++ms) {
#pragma unroll
      for (int r = 0; r < 8; ++r) {
        int m = m0w + ms * 16 + kh * 8 + r;
        Zb[(size_t)m * HH + n] = (bf16_t)tanhf(acc[ms][j][r] + bias);
      }
    }
  }
}

// ---------------------------------------------------------------------------
// f_node layer 2 + RK4 update. k = Z @ W2b^T + b2.
//  stage 1: acc = k;    y = ht + 0.5*dt*k
//  stage 2: acc += 2k;  y = ht + 0.5*dt*k
//  stage 3: acc += 2k;  y = ht + dt*k
//  stage 4: acc += k;   h_next = ht + (dt/6)*acc
// ---------------------------------------------------------------------------
__global__ __launch_bounds__(128) void k_fnode_l2(
    const bf16_t* __restrict__ Zb, const bf16_t* __restrict__ W2b,
    const float* __restrict__ b2, const float* __restrict__ ht32,
    float* __restrict__ accbuf, bf16_t* __restrict__ Yb, float* __restrict__ hnext,
    const float* __restrict__ t0p, const float* __restrict__ t1p, int stage) {
  __shared__ __align__(32) bf16_t smem[2][LROWS * LSTRIDE];

  const int tid  = threadIdx.x;
  const int wv   = tid >> 5;
  const int lane = tid & 31;
  const int nl   = lane & 15;
  const int kh   = lane >> 4;

  const int TG   = HH / 64;
  const int ng   = (blockIdx.x % TG) * 64;
  const int m0w  = (blockIdx.x / TG) * 128 + wv * 32;
  const int rowA0 = m0w + nl;
  const int rowA1 = m0w + 16 + nl;

  const int sr = tid >> 1, sh = tid & 1;
  const int wrow = ng + sr;

  v8f acc[2][4] = {};
  v16bf breg = load_wslice(W2b, HH, wrow, 0, sh);
  stage_store(smem[0], sr, sh, breg);
  __syncthreads();
  for (int i = 0; i < HH / 32; ++i) {
    const int cur = i & 1;
    const bool more = (i + 1) < (HH / 32);
    if (more) breg = load_wslice(W2b, HH, wrow, (i + 1) * 32, sh);
    const int k0 = i * 32;
    v16bf a0 = load_a_frag(Zb, HH, rowA0, k0, kh);
    v16bf a1 = load_a_frag(Zb, HH, rowA1, k0, kh);
#pragma unroll
    for (int j = 0; j < 4; ++j) {
      v16bf b = lds_b_frag(smem[cur], j * 16 + nl, kh);
      acc[0][j] = WMMA_BF16(a0, b, acc[0][j]);
      acc[1][j] = WMMA_BF16(a1, b, acc[1][j]);
    }
    if (more) stage_store(smem[cur ^ 1], sr, sh, breg);
    __syncthreads();
  }

  const float dt = t1p[0] - t0p[0];
#pragma unroll
  for (int j = 0; j < 4; ++j) {
    int n = ng + j * 16 + nl;
    float bias = b2[n];
#pragma unroll
    for (int ms = 0; ms < 2; ++ms) {
#pragma unroll
      for (int r = 0; r < 8; ++r) {
        int m = m0w + ms * 16 + kh * 8 + r;
        size_t idx = (size_t)m * HH + n;
        float kv = acc[ms][j][r] + bias;
        float ht = ht32[idx];
        if (stage == 1) {
          accbuf[idx] = kv;
          Yb[idx] = (bf16_t)(ht + 0.5f * dt * kv);
        } else if (stage == 2) {
          accbuf[idx] += 2.0f * kv;
          Yb[idx] = (bf16_t)(ht + 0.5f * dt * kv);
        } else if (stage == 3) {
          accbuf[idx] += 2.0f * kv;
          Yb[idx] = (bf16_t)(ht + dt * kv);
        } else {
          float a4 = accbuf[idx] + kv;
          hnext[idx] = ht + (dt / 6.0f) * a4;
        }
      }
    }
  }
}

// ---------------------------------------------------------------------------
extern "C" void kernel_launch(void* const* d_in, const int* in_sizes, int n_in,
                              void* d_out, int out_size, void* d_ws, size_t ws_size,
                              hipStream_t stream) {
  (void)in_sizes; (void)n_in; (void)out_size; (void)ws_size;
  const float* x    = (const float*)d_in[0];
  const float* h    = (const float*)d_in[1];
  const float* c    = (const float*)d_in[2];
  const float* t0   = (const float*)d_in[3];
  const float* t1   = (const float*)d_in[4];
  const float* W_ih = (const float*)d_in[5];
  const float* W_hh = (const float*)d_in[6];
  const float* b_ih = (const float*)d_in[7];
  const float* b_hh = (const float*)d_in[8];
  const float* W1   = (const float*)d_in[9];
  const float* b1   = (const float*)d_in[10];
  const float* W2   = (const float*)d_in[11];
  const float* b2   = (const float*)d_in[12];

  float* out_h = (float*)d_out;                 // h_next [B,H]
  float* out_c = out_h + (size_t)BB * HH;       // c_next [B,H]

  char* ws = (char*)d_ws;
  size_t off = 0;
  auto alloc = [&](size_t bytes) -> void* {
    void* p = ws + off;
    off = (off + bytes + 255) & ~(size_t)255;
    return p;
  };
  bf16_t* xb    = (bf16_t*)alloc((size_t)BB * II * 2);
  bf16_t* hb    = (bf16_t*)alloc((size_t)BB * HH * 2);
  bf16_t* wihb  = (bf16_t*)alloc((size_t)4 * HH * II * 2);
  bf16_t* whhb  = (bf16_t*)alloc((size_t)4 * HH * HH * 2);
  bf16_t* w1b   = (bf16_t*)alloc((size_t)HH * HH * 2);
  bf16_t* w2b   = (bf16_t*)alloc((size_t)HH * HH * 2);
  float*  bsum  = (float*)alloc((size_t)4 * HH * 4);
  float*  b1eff = (float*)alloc((size_t)3 * HH * 4);
  float*  ht32  = (float*)alloc((size_t)BB * HH * 4);
  bf16_t* htb   = (bf16_t*)alloc((size_t)BB * HH * 2);
  bf16_t* zb    = (bf16_t*)alloc((size_t)BB * HH * 2);
  bf16_t* yb    = (bf16_t*)alloc((size_t)BB * HH * 2);
  float*  accb  = (float*)alloc((size_t)BB * HH * 4);

  // Preparation: bf16 conversions + bias folding
  k_cvt_bf16<<<1024, 256, 0, stream>>>(x,    xb,   BB * II);
  k_cvt_bf16<<<2048, 256, 0, stream>>>(h,    hb,   BB * HH);
  k_cvt_bf16<<<1024, 256, 0, stream>>>(W_ih, wihb, 4 * HH * II);
  k_cvt_bf16<<<2048, 256, 0, stream>>>(W_hh, whhb, 4 * HH * HH);
  k_cvt_bf16<<<1024, 256, 0, stream>>>(W2,   w2b,  HH * HH);
  k_cvt_W1  <<<1024, 256, 0, stream>>>(W1,   w1b);
  k_bias    <<<(4 * HH + 255) / 256, 256, 0, stream>>>(b_ih, b_hh, b1, W1, t0, t1,
                                                       bsum, b1eff);

  // Fused LSTM cell (gates GEMM + pointwise): produces c_next, h_tilde (f32 + bf16)
  const int gblocks = (BB / 128) * (HH / 16);   // 4096
  k_gates<<<gblocks, 128, 0, stream>>>(xb, hb, wihb, whhb, bsum, c, out_c, ht32, htb);

  // RK4: 4 stages of f_node (two WMMA GEMMs each), fully fused updates
  const int fblocks = (BB / 128) * (HH / 64);   // 1024
  k_fnode_l1<<<fblocks, 128, 0, stream>>>(htb, w1b, b1eff + 0 * HH, zb);
  k_fnode_l2<<<fblocks, 128, 0, stream>>>(zb, w2b, b2, ht32, accb, yb, out_h, t0, t1, 1);
  k_fnode_l1<<<fblocks, 128, 0, stream>>>(yb,  w1b, b1eff + 1 * HH, zb);
  k_fnode_l2<<<fblocks, 128, 0, stream>>>(zb, w2b, b2, ht32, accb, yb, out_h, t0, t1, 2);
  k_fnode_l1<<<fblocks, 128, 0, stream>>>(yb,  w1b, b1eff + 1 * HH, zb);
  k_fnode_l2<<<fblocks, 128, 0, stream>>>(zb, w2b, b2, ht32, accb, yb, out_h, t0, t1, 3);
  k_fnode_l1<<<fblocks, 128, 0, stream>>>(yb,  w1b, b1eff + 2 * HH, zb);
  k_fnode_l2<<<fblocks, 128, 0, stream>>>(zb, w2b, b2, ht32, accb, yb, out_h, t0, t1, 4);
}
